// DM_35210141892754
// MI455X (gfx1250) — matrix operand order repaired and checked by reading, not compiled, not run on hardware
//
#include <hip/hip_runtime.h>
#include <hip/hip_bf16.h>
#include <math.h>

// ---------------------------------------------------------------------------
// Problem constants (from reference setup_inputs; step_size=2, so L=8192)
// ---------------------------------------------------------------------------
#define BB      2        // batch
#define CC      64       // channels
#define HIMG    128
#define WIMG    128
#define LSEQ    8192     // (HIMG * 2*WIMG) / (s*s), s=2
#define DINNER  128
#define DSTATE  16
#define DTRANK  4
#define DBL_LD  48       // padded row stride for (dt:4 | B:16 | C:16) = 36 -> 48
#define KDT_PAD 32       // dt_proj K padded 4 -> 32 (zero-filled weights)
#define NXP_PAD 48       // x_proj N padded 36 -> 48 (zero-filled weights)

typedef __attribute__((ext_vector_type(16))) _Float16 v16h;
typedef __attribute__((ext_vector_type(8)))  float    v8f;
// 32-byte vector of halfs with relaxed (16B) alignment: lowers to 2x b128 loads
typedef v16h v16h_a __attribute__((aligned(16)));

// OFF[i] = (r, l, b, t) pairs from the reference
__device__ __constant__ int OFFT[4][4][2] = {
  {{1,0},{0,0},{0,1},{1,1}},
  {{0,0},{1,0},{1,1},{0,1}},
  {{0,1},{1,1},{1,0},{0,0}},
  {{1,1},{0,1},{0,0},{1,0}},
};

__device__ __forceinline__ float siluf(float x) { return x / (1.f + __expf(-x)); }

// ---------------------------------------------------------------------------
// Weight prep: convert all weights to f16 once per launch (zero-padding the
// two awkward shapes so every GEMM is guard-free).
// ---------------------------------------------------------------------------
__global__ void cvt_f16_kernel(const float* __restrict__ src,
                               _Float16* __restrict__ dst, int n)
{
  int i = blockIdx.x * blockDim.x + threadIdx.x;
  if (i < n) dst[i] = (_Float16)src[i];
}

__global__ void pad_xproj_kernel(const float* __restrict__ W,  // (4,36,128)
                                 _Float16* __restrict__ Wp)    // (4,48,128)
{
  int idx = blockIdx.x * blockDim.x + threadIdx.x;
  if (idx >= 4 * NXP_PAD * DINNER) return;
  int k = idx & 127;
  int n = (idx >> 7) % NXP_PAD;
  int d = idx / (NXP_PAD * DINNER);
  Wp[idx] = (n < 36) ? (_Float16)W[((size_t)d * 36 + n) * DINNER + k]
                     : (_Float16)0.f;
}

__global__ void pad_dtw_kernel(const float* __restrict__ W,    // (4,128,4)
                               _Float16* __restrict__ Wp)      // (4,128,32)
{
  int idx = blockIdx.x * blockDim.x + threadIdx.x;
  if (idx >= 4 * DINNER * KDT_PAD) return;
  int k = idx & 31;
  int n = (idx >> 5) & 127;
  int d = idx >> 12;
  Wp[idx] = (k < 4) ? (_Float16)W[((size_t)d * DINNER + n) * 4 + k]
                    : (_Float16)0.f;
}

// ---------------------------------------------------------------------------
// RCDS gather: build feats (B,4,L,C) from ref/src feature maps.
// ---------------------------------------------------------------------------
__global__ void rcds_kernel(const float* __restrict__ f0,
                            const float* __restrict__ f1,
                            const int* __restrict__ srcnum,
                            float* __restrict__ feats)
{
  int idx = blockIdx.x * blockDim.x + threadIdx.x;      // over B*4*L*C
  if (idx >= BB * 4 * LSEQ * CC) return;
  int c = idx & 63;
  int l = (idx >> 6) & (LSEQ - 1);
  int k = (idx >> 19) & 3;
  int b = idx >> 21;
  int i = ((srcnum[0] % 4) + 4) % 4;
  int o0 = OFFT[i][k][0], o1 = OFFT[i][k][1];

  float v = 0.f;
  const size_t plane = (size_t)(b * CC + c) * (HIMG * WIMG);
  if (k == 0) {                        // fr = [f0|f1] width, col-major tiles
    int a = l & 63, b2 = l >> 6;
    int row = o0 + 2 * a, gcol = o1 + 2 * b2;
    v = (gcol < WIMG) ? f0[plane + (size_t)row * WIMG + gcol]
                      : f1[plane + (size_t)row * WIMG + (gcol - WIMG)];
  } else if (k == 1) {                 // fl = [f1|f0] width, reversed
    int flat = LSEQ - 1 - l;
    int a = flat & 63, b2 = flat >> 6;
    int row = o0 + 2 * a, gcol = o1 + 2 * b2;
    v = (gcol < WIMG) ? f1[plane + (size_t)row * WIMG + gcol]
                      : f0[plane + (size_t)row * WIMG + (gcol - WIMG)];
  } else if (k == 2) {                 // fb = [f0;f1] height, row-major tiles
    int a = l >> 6, b2 = l & 63;
    int grow = o0 + 2 * a, col = o1 + 2 * b2;
    v = (grow < HIMG) ? f0[plane + (size_t)grow * WIMG + col]
                      : f1[plane + (size_t)(grow - HIMG) * WIMG + col];
  } else {                             // ft = [f1;f0] height, reversed
    int flat = LSEQ - 1 - l;
    int a = flat >> 6, b2 = flat & 63;
    int grow = o0 + 2 * a, col = o1 + 2 * b2;
    v = (grow < HIMG) ? f1[plane + (size_t)grow * WIMG + col]
                      : f0[plane + (size_t)(grow - HIMG) * WIMG + col];
  }
  feats[idx] = v;
}

// ---------------------------------------------------------------------------
// LayerNorm over C=64.
// ---------------------------------------------------------------------------
__global__ void ln_kernel(const float* __restrict__ in,
                          const float* __restrict__ w,
                          const float* __restrict__ bvec,
                          float* __restrict__ out, int nTok, int d)
{
  int t = blockIdx.x * blockDim.x + threadIdx.x;
  if (t >= nTok) return;
  size_t row;
  if (d >= 0) {
    int b = t / LSEQ, l = t % LSEQ;
    row = ((size_t)(b * 4 + d) * LSEQ + l);
  } else {
    row = (size_t)t;
  }
  const float* x = in + row * CC;
  float mu = 0.f;
  #pragma unroll 8
  for (int c = 0; c < CC; ++c) mu += x[c];
  mu *= (1.f / CC);
  float var = 0.f;
  #pragma unroll 8
  for (int c = 0; c < CC; ++c) { float dv = x[c] - mu; var += dv * dv; }
  var *= (1.f / CC);
  float inv = rsqrtf(var + 1e-5f);
  float* o = out + (size_t)t * CC;
  #pragma unroll 8
  for (int c = 0; c < CC; ++c) o[c] = (x[c] - mu) * inv * w[c] + bvec[c];
}

// ---------------------------------------------------------------------------
// WMMA GEMM, guard-free:  D[m, n] = sum_k A[m,k] * W16[n,k]  (+ epilogue)
// Launcher guarantees: M % 16 == 0, N % (16*NT) == 0, K % 32 == 0,
// lda % 4 == 0, all fragment loads 16B-aligned.
// One wave computes a 16 x (16*NT) strip: one A fragment feeds NT WMMAs.
// A: f32 activations, cvt'd in-register; W16: pre-converted f16 weights,
// loaded directly as two b128 per fragment (zero conversion VALU).
// ---------------------------------------------------------------------------
#define EPI_NONE          0
#define EPI_BIAS_SOFTPLUS 1
#define EPI_GELU          2
#define EPI_RESADD        3
#define EPI_ACCUM         4

template <int NT>
__global__ __launch_bounds__(32)
void gemm_wmma_kernel(const float* __restrict__ A, int lda, long long strideA,
                      const _Float16* __restrict__ W16,
                      const float* __restrict__ bias,
                      const float* __restrict__ Res, int ldr, long long strideR,
                      float* __restrict__ C, int ldc, long long strideC,
                      int K, int mode)
{
  const int lane = threadIdx.x;           // 0..31, wave32
  const int half = lane >> 4;
  const int lid  = lane & 15;
  const int nBase = blockIdx.x * (16 * NT);
  const int tileM = blockIdx.y;
  const int bz    = blockIdx.z;

  const float* Ab = A + (size_t)bz * strideA + (size_t)(tileM * 16 + lid) * lda;
  float*       Cb = C + (size_t)bz * strideC;
  const float* Rb = Res ? (Res + (size_t)bz * strideR) : nullptr;

  v8f acc[NT];
  #pragma unroll
  for (int t = 0; t < NT; ++t) acc[t] = (v8f){0.f,0.f,0.f,0.f,0.f,0.f,0.f,0.f};

  for (int k0 = 0; k0 < K; k0 += 32) {
    // A fragment (contiguous k order, not the ISA table): lanes 0-15 rows M=lid,
    // K = k0..k0+15; lanes 16-31 K = k0+16..k0+31 (matches B).
    const float4* apLo = (const float4*)(Ab + k0 + half * 16);
    const float4* apHi = (const float4*)(Ab + k0 + half * 16 + 8);
    float4 a0 = apLo[0], a1 = apLo[1];
    float4 a2 = apHi[0], a3 = apHi[1];
    v16h af;
    af[0]  = (_Float16)a0.x; af[1]  = (_Float16)a0.y;
    af[2]  = (_Float16)a0.z; af[3]  = (_Float16)a0.w;
    af[4]  = (_Float16)a1.x; af[5]  = (_Float16)a1.y;
    af[6]  = (_Float16)a1.z; af[7]  = (_Float16)a1.w;
    af[8]  = (_Float16)a2.x; af[9]  = (_Float16)a2.y;
    af[10] = (_Float16)a2.z; af[11] = (_Float16)a2.w;
    af[12] = (_Float16)a3.x; af[13] = (_Float16)a3.y;
    af[14] = (_Float16)a3.z; af[15] = (_Float16)a3.w;

    #pragma unroll
    for (int t = 0; t < NT; ++t) {
      // B fragment (32x16): lanes 0-15 col n=lid K=k0..k0+15;
      // lanes 16-31 same col, K=k0+16..31. 16 halfs = 2x b128 loads.
      const _Float16* Wr =
          W16 + (size_t)(nBase + t * 16 + lid) * K + k0 + half * 16;
      v16h bf = *(const v16h_a*)Wr;
      acc[t] = __builtin_amdgcn_wmma_f32_16x16x32_f16(
          false, af, false, bf, (short)0, acc[t], false, false);
    }
  }

  // C/D layout: lanes 0-15: N=lid, M=r; lanes 16-31: N=lid, M=r+8.
  #pragma unroll
  for (int t = 0; t < NT; ++t) {
    const int n = nBase + t * 16 + lid;
    #pragma unroll
    for (int r = 0; r < 8; ++r) {
      const int m = tileM * 16 + half * 8 + r;
      float v = acc[t][r];
      if (mode == EPI_BIAS_SOFTPLUS) {
        v += bias[n];
        v = (v > 20.f) ? v : log1pf(__expf(v));          // softplus
      } else if (mode == EPI_GELU) {
        v = 0.5f * v * (1.f + erff(v * 0.70710678118f)); // exact GELU
      } else if (mode == EPI_RESADD) {
        v += Rb[(size_t)m * ldr + n];
      }
      if (mode == EPI_ACCUM) Cb[(size_t)m * ldc + n] += v;
      else                   Cb[(size_t)m * ldc + n]  = v;
    }
  }
}

// ---------------------------------------------------------------------------
// Causal depthwise conv (D_CONV=4) + bias + SiLU.
// ---------------------------------------------------------------------------
__global__ void conv_silu_kernel(const float* __restrict__ xz,
                                 const float* __restrict__ cw,   // [128,4]
                                 const float* __restrict__ cb,   // [128]
                                 float* __restrict__ xh)
{
  int idx = blockIdx.x * blockDim.x + threadIdx.x;   // over B*L*128
  if (idx >= BB * LSEQ * DINNER) return;
  int ch = idx & 127;
  int t  = (idx >> 7) & (LSEQ - 1);
  int b  = idx >> 20;
  float s = cb[ch];
  #pragma unroll
  for (int j = 0; j < 4; ++j) {
    int tt = t - 3 + j;
    if (tt >= 0) s += cw[ch * 4 + j] * xz[((size_t)(b * LSEQ + tt)) * 256 + ch];
  }
  xh[(size_t)(b * LSEQ + t) * DINNER + ch] = siluf(s);
}

// ---------------------------------------------------------------------------
// Selective scan: one lane per (channel, state) chain (4096 chains).
// Block 256 = 16 channels x 16 states; grid = B * 8. 16-lane shfl_xor tree
// contracts h·C per step; fused skip + SiLU gate.
// ---------------------------------------------------------------------------
__global__ __launch_bounds__(256)
void scan_kernel(const float* __restrict__ dbl,    // (B,L,48): [dt4 | B16 | C16]
                 const float* __restrict__ dt,     // (B,L,128) after softplus
                 const float* __restrict__ xh,     // (B,L,128)
                 const float* __restrict__ xz,     // (B,L,256), z at +128
                 const float* __restrict__ A_log,  // [128,16] (pre-offset by d)
                 const float* __restrict__ Dp,     // [128]
                 float* __restrict__ y)            // (B,L,128)
{
  const int tid = threadIdx.x;
  const int s   = tid & 15;
  const int chL = tid >> 4;
  const int b   = blockIdx.x >> 3;
  const int ch  = ((blockIdx.x & 7) << 4) + chL;

  const float Ac  = -__expf(A_log[ch * DSTATE + s]);
  const float dsk = Dp[ch];
  float h = 0.f;

  for (int t = 0; t < LSEQ; ++t) {
    const size_t row = (size_t)(b * LSEQ + t);
    float dtv = dt[row * DINNER + ch];
    float xv  = xh[row * DINNER + ch];
    float Bv  = dbl[row * DBL_LD + DTRANK + s];
    float Cv  = dbl[row * DBL_LD + DTRANK + DSTATE + s];
    h = __expf(dtv * Ac) * h + dtv * xv * Bv;
    float p = h * Cv;
    p += __shfl_xor(p, 8, 32);
    p += __shfl_xor(p, 4, 32);
    p += __shfl_xor(p, 2, 32);
    p += __shfl_xor(p, 1, 32);
    if (s == 0) {
      float zv = xz[row * 256 + DINNER + ch];
      y[row * DINNER + ch] = (p + xv * dsk) * siluf(zv);
    }
  }
}

// ---------------------------------------------------------------------------
// Merge: gather the 4 directional streams back into (d0, d1) (B,C,H,W) each.
// ---------------------------------------------------------------------------
__global__ void merge_kernel(const float* __restrict__ F,   // feats (B,4,L,C)
                             const int* __restrict__ srcnum,
                             float* __restrict__ out)
{
  int idx = blockIdx.x * blockDim.x + threadIdx.x;  // over 2*B*C*H*W
  if (idx >= 2 * BB * CC * HIMG * WIMG) return;
  int w = idx & 127;
  int h = (idx >> 7) & 127;
  int c = (idx >> 14) & 63;
  int b = (idx >> 20) & 1;
  int which = idx >> 21;
  int i = ((srcnum[0] % 4) + 4) % 4;

  float acc = 0.f;
  {
    int o0 = OFFT[i][0][0], o1 = OFFT[i][0][1];
    if ((h & 1) == o0 && (w & 1) == o1) {
      int a = (h - o0) >> 1;
      int b2 = ((w + which * WIMG) - o1) >> 1;
      int l = b2 * 64 + a;
      acc += F[((size_t)(b * 4 + 0) * LSEQ + l) * CC + c];
    }
  }
  {
    int o0 = OFFT[i][1][0], o1 = OFFT[i][1][1];
    if ((h & 1) == o0 && (w & 1) == o1) {
      int a = (h - o0) >> 1;
      int b2 = ((w + (1 - which) * WIMG) - o1) >> 1;
      int l = (LSEQ - 1) - (b2 * 64 + a);
      acc += F[((size_t)(b * 4 + 1) * LSEQ + l) * CC + c];
    }
  }
  {
    int o0 = OFFT[i][2][0], o1 = OFFT[i][2][1];
    if ((h & 1) == o0 && (w & 1) == o1) {
      int a = ((h + which * HIMG) - o0) >> 1;
      int b2 = (w - o1) >> 1;
      int l = a * 64 + b2;
      acc += F[((size_t)(b * 4 + 2) * LSEQ + l) * CC + c];
    }
  }
  {
    int o0 = OFFT[i][3][0], o1 = OFFT[i][3][1];
    if ((h & 1) == o0 && (w & 1) == o1) {
      int a = ((h + (1 - which) * HIMG) - o0) >> 1;
      int b2 = (w - o1) >> 1;
      int l = (LSEQ - 1) - (a * 64 + b2);
      acc += F[((size_t)(b * 4 + 3) * LSEQ + l) * CC + c];
    }
  }
  out[idx] = acc;
}

// ---------------------------------------------------------------------------
// Orchestration. Workspace (f32 region then f16 weight region), ~83 MB total
// (fits in the 192 MB L2):
//   feats  : 4,194,304   feats2 : 4,194,304   xln : 1,048,576
//   xz     : 4,194,304   xh     : 2,097,152   dbl : 786,432
//   dtb    : 2,097,152   yb     : 2,097,152
//   f16 weights: in(65536) xp(24576) dt(16384) out(32768) fc1(16384) fc2(16384)
// ---------------------------------------------------------------------------
extern "C" void kernel_launch(void* const* d_in, const int* in_sizes, int n_in,
                              void* d_out, int out_size, void* d_ws, size_t ws_size,
                              hipStream_t stream)
{
  const float* ref      = (const float*)d_in[0];
  const float* src      = (const float*)d_in[1];
  const float* norm_w   = (const float*)d_in[2];
  const float* norm_b   = (const float*)d_in[3];
  const float* in_proj  = (const float*)d_in[4];   // (4,256,64)
  const float* conv_w   = (const float*)d_in[5];   // (4,128,4)
  const float* conv_b   = (const float*)d_in[6];   // (4,128)
  const float* x_proj   = (const float*)d_in[7];   // (4,36,128)
  const float* dt_w     = (const float*)d_in[8];   // (4,128,4)
  const float* dt_b     = (const float*)d_in[9];   // (4,128)
  const float* A_log    = (const float*)d_in[10];  // (4,128,16)
  const float* D_skip   = (const float*)d_in[11];  // (4,128)
  const float* out_proj = (const float*)d_in[12];  // (4,64,128)
  const float* fnorm_w  = (const float*)d_in[13];
  const float* fnorm_b  = (const float*)d_in[14];
  const float* fc1_w    = (const float*)d_in[15];  // (256,64)
  const float* fc2_w    = (const float*)d_in[16];  // (64,256)
  const int*   src_num  = (const int*)d_in[18];

  float* ws     = (float*)d_ws;
  float* feats  = ws;
  float* feats2 = ws + 4194304;
  float* xln    = ws + 8388608;
  float* xz     = ws + 9437184;
  float* xh     = ws + 13631488;
  float* dbl    = ws + 15728640;
  float* dtb    = ws + 16515072;
  float* yb     = ws + 18612224;

  _Float16* h16   = (_Float16*)(ws + 20709376);
  _Float16* h_in  = h16;                 // 4*256*64  = 65536
  _Float16* h_xp  = h16 + 65536;         // 4*48*128  = 24576 (padded)
  _Float16* h_dt  = h16 + 90112;         // 4*128*32  = 16384 (padded)
  _Float16* h_out = h16 + 106496;        // 4*64*128  = 32768
  _Float16* h_fc1 = h16 + 139264;        // 256*64    = 16384
  _Float16* h_fc2 = h16 + 155648;        // 64*256    = 16384

  const long long sA_tok = (long long)LSEQ * CC;
  const long long sA_xz  = (long long)LSEQ * 256;
  const long long sA_di  = (long long)LSEQ * DINNER;
  const long long sA_dbl = (long long)LSEQ * DBL_LD;
  const long long sFeat  = (long long)4 * LSEQ * CC;

  // 0) Weight prep: f32 -> f16 (with zero-padding for x_proj / dt_proj)
  cvt_f16_kernel<<<(65536 + 255) / 256, 256, 0, stream>>>(in_proj, h_in, 65536);
  pad_xproj_kernel<<<(4 * NXP_PAD * DINNER + 255) / 256, 256, 0, stream>>>(x_proj, h_xp);
  pad_dtw_kernel<<<(4 * DINNER * KDT_PAD + 255) / 256, 256, 0, stream>>>(dt_w, h_dt);
  cvt_f16_kernel<<<(32768 + 255) / 256, 256, 0, stream>>>(out_proj, h_out, 32768);
  cvt_f16_kernel<<<(16384 + 255) / 256, 256, 0, stream>>>(fc1_w, h_fc1, 16384);
  cvt_f16_kernel<<<(16384 + 255) / 256, 256, 0, stream>>>(fc2_w, h_fc2, 16384);

  // 1) RCDS gather
  rcds_kernel<<<(BB * 4 * LSEQ * CC) / 256, 256, 0, stream>>>(ref, src, src_num, feats);

  // 2) Four directional Mamba blocks
  for (int d = 0; d < 4; ++d) {
    ln_kernel<<<(BB * LSEQ + 255) / 256, 256, 0, stream>>>(
        feats, norm_w + d * CC, norm_b + d * CC, xln, BB * LSEQ, d);

    // in_proj: (L,64) x (256,64)^T -> (L,256). N=256, NT=4.
    dim3 g1(256 / 64, LSEQ / 16, BB);
    gemm_wmma_kernel<4><<<g1, 32, 0, stream>>>(
        xln, CC, sA_tok, h_in + (size_t)d * 256 * 64,
        nullptr, nullptr, 0, 0, xz, 256, sA_xz, CC, EPI_NONE);

    conv_silu_kernel<<<(BB * LSEQ * DINNER + 255) / 256, 256, 0, stream>>>(
        xz, conv_w + d * DINNER * 4, conv_b + d * DINNER, xh);

    // x_proj (padded): (L,128) x (48,128)^T -> (L,48). N=48, NT=3.
    dim3 g2(1, LSEQ / 16, BB);
    gemm_wmma_kernel<3><<<g2, 32, 0, stream>>>(
        xh, DINNER, sA_di, h_xp + (size_t)d * NXP_PAD * DINNER,
        nullptr, nullptr, 0, 0, dbl, DBL_LD, sA_dbl, DINNER, EPI_NONE);

    // dt_proj (K padded to 32; zero weight cols kill dbl's B/C columns):
    dim3 g3(128 / 64, LSEQ / 16, BB);
    gemm_wmma_kernel<4><<<g3, 32, 0, stream>>>(
        dbl, DBL_LD, sA_dbl, h_dt + (size_t)d * DINNER * KDT_PAD,
        dt_b + d * DINNER, nullptr, 0, 0,
        dtb, DINNER, sA_di, KDT_PAD, EPI_BIAS_SOFTPLUS);

    // selective scan + skip + gate
    scan_kernel<<<BB * (DINNER / 16), 256, 0, stream>>>(
        dbl, dtb, xh, xz, A_log + (size_t)d * DINNER * DSTATE,
        D_skip + d * DINNER, yb);

    // out_proj + residual: feats2[b,d] = y @ W^T + feats[b,d]. N=64, NT=4.
    dim3 g4(64 / 64, LSEQ / 16, BB);
    gemm_wmma_kernel<4><<<g4, 32, 0, stream>>>(
        yb, DINNER, sA_di, h_out + (size_t)d * 64 * 128,
        nullptr, feats + (size_t)d * LSEQ * CC, CC, sFeat,
        feats2 + (size_t)d * LSEQ * CC, CC, sFeat, DINNER, EPI_RESADD);
  }

  // 3) Final LN (into feats, reused as hln) + MLP accumulated in place
  ln_kernel<<<(BB * 4 * LSEQ + 255) / 256, 256, 0, stream>>>(
      feats2, fnorm_w, fnorm_b, feats, BB * 4 * LSEQ, -1);

  for (int bk = 0; bk < BB * 4; ++bk) {            // 8 chunks of 8192 tokens
    const float* Ab = feats + (size_t)bk * LSEQ * CC;
    dim3 g5(256 / 64, LSEQ / 16, 1);
    gemm_wmma_kernel<4><<<g5, 32, 0, stream>>>(
        Ab, CC, 0, h_fc1, nullptr, nullptr, 0, 0,
        xz, 256, 0, CC, EPI_GELU);
    dim3 g6(64 / 64, LSEQ / 16, 1);
    gemm_wmma_kernel<4><<<g6, 32, 0, stream>>>(
        xz, 256, 0, h_fc2, nullptr, nullptr, 0, 0,
        feats2 + (size_t)bk * LSEQ * CC, CC, 0, 256, EPI_ACCUM);
  }

  // 4) Merge back to (d0, d1)
  merge_kernel<<<(2 * BB * CC * HIMG * WIMG) / 256, 256, 0, stream>>>(
      feats2, src_num, (float*)d_out);
}
